// Sins_32109175505143
// MI455X (gfx1250) — matrix operand order repaired and checked by reading, not compiled, not run on hardware
//
#include <hip/hip_runtime.h>
#include <hip/hip_bf16.h>
#include <stdint.h>

// ---------------------------------------------------------------------------
// DDSP "Sins" synthesizer forward pass for MI455X (gfx1250, wave32, WMMA).
//
// Heavy math:
//   * k_gemm_f16  -- V_WMMA_F32_16X16X32_F16; 4 waves/WG compute a 64x16 C
//                    macro-tile; the shared Bt (16x32 f16) K-step tile is
//                    staged into LDS with GLOBAL_LOAD_ASYNC_TO_LDS_B128
//                    (ASYNCcnt + s_wait_asynccnt + barrier), B fragments are
//                    then read via ds_load_b128. A fragments stream from
//                    global as clause-paired global_load_b128.
//   * k_synth     -- 134M v_sin_f32 harmonic bank (transcendental bound).
//   * k_ir_* / k_filter_out -- frequency_filter as direct cosine-DFT IR +
//                    time-domain conv (== 1024-pt FFT overlap-add since the
//                    linear conv length 1021 <= 1024).
//
// d_in layout assumption (jax pytree, dicts sorted by key):
//  0 units 1 f0 2 vol 3 conv1_b 4 conv1_w 5 conv2_b 6 conv2_w 7 f0_b 8 f0_w
//  9 fin_b 10 fin_s 11 gn_b 12 gn_s
//  13+14*l : cln_b cln_s dw_b dw_w ln1_b ln1_s o_b o_w pw1_b pw1_w pw2_b pw2_w qkv_b qkv_w
//  55 out_b 56 out_w 57 ph_b 58 ph_w 59 vol_b 60 vol_w
// d_out: [signal BT][hidden BF*D][harmonic BT][noise BT] (floats)
// d_ws: ~72MB, offsets below.
// ---------------------------------------------------------------------------

typedef _Float16 f16;
typedef __attribute__((ext_vector_type(16))) _Float16 v16h;
typedef __attribute__((ext_vector_type(8)))  float    v8f;

#define FF      512          // frames
#define BB      4            // batch
#define BLK     512          // samples per frame
#define TT      (FF*BLK)     // 262144 samples per batch
#define BTOT    (BB*TT)      // 1048576
#define BF      (BB*FF)      // 2048 rows
#define DM      256          // model dim
#define NHARM   128
#define LFILT   510          // 2*(M-1), M=256
#define PI_F    3.14159265358979323846f

// ---------------------------------------------------------------- WMMA GEMM
// C[m,n] = alpha * sum_k A[m,k]*Bt[n,k] + bias[n] (+ Cres[m,n])
// Block: 128 threads (4 waves) -> 64x16 C macro-tile (wave w owns rows
// [mT*64+w*16, +16)). Bt K-tile (16 cols x 32 k = 1KB) staged in LDS once per
// K-step by waves 0-1 via async-to-LDS, consumed by all 4 waves.
// Batched over grid.z: z -> (zb=z/H, zh=z%H) pointer offsets.
__global__ __launch_bounds__(128)
void k_gemm_f16(const f16* A, const f16* Bt, const float* bias,
                const float* Cres, float* C,
                int M, int N, int K, int lda, int ldb, int ldc,
                float alpha, int H,
                long aOffB, long aOffH, long bOffB, long bOffH,
                long cOffB, long cOffH)
{
    int z  = blockIdx.z;
    int zb = z / H, zh = z % H;
    A  += zb * aOffB + zh * aOffH;
    Bt += zb * bOffB + zh * bOffH;
    C  += zb * cOffB + zh * cOffH;
    if (Cres) Cres += zb * cOffB + zh * cOffH;

    __shared__ f16 sB[16 * 32];                 // [col 0..15][k 0..31]

    const int nT   = blockIdx.x, mT = blockIdx.y;
    const int wave = threadIdx.x >> 5;
    const int lane = threadIdx.x & 31;
    const int half = lane >> 4;                 // lanes 0-15 / 16-31
    const int l16  = lane & 15;
    const int mRow = mT * 64 + wave * 16 + l16; // A fragment row for this lane
    const int nColG = nT * 16;                  // first output column of tile

    v8f acc = {0.f,0.f,0.f,0.f,0.f,0.f,0.f,0.f};

    for (int k0 = 0; k0 < K; k0 += 32) {
        __syncthreads();                        // prior DS reads done
        if (threadIdx.x < 64) {                 // waves 0,1: 64 x 16B chunks
            int row = threadIdx.x >> 2;         // Bt row (output column) 0..15
            int seg = threadIdx.x & 3;          // 8-element k segment
            const f16* gp = Bt + (long)(nColG + row) * ldb + k0 + seg * 8;
            unsigned lp = (unsigned)(uintptr_t)(&sB[row * 32 + seg * 8]);
            asm volatile("global_load_async_to_lds_b128 %0, %1, off"
                         :: "v"(lp), "v"((unsigned long long)(uintptr_t)gp)
                         : "memory");
        }
        asm volatile("s_wait_asynccnt 0" ::: "memory");
        __syncthreads();                        // LDS tile visible to all waves

        v16h af, bf;
        // A 16x32 f16 layout: lanes<16 hold K {0..7,16..23}, lanes>=16 {8..15,24..31}
        const f16* ap = A + (long)mRow * lda + k0 + half * 8;
        #pragma unroll
        for (int i = 0; i < 8; ++i) { af[i] = ap[i]; af[8 + i] = ap[16 + i]; }
        // B 32x16 f16 layout from LDS: lane = column l16; lanes<16 K {0..7,16..23}, >=16 {8..15,24..31}
        const f16* bp = &sB[l16 * 32 + half * 8];
        #pragma unroll
        for (int i = 0; i < 8; ++i) { bf[i] = bp[i]; bf[8 + i] = bp[16 + i]; }
        acc = __builtin_amdgcn_wmma_f32_16x16x32_f16(false, af, false, bf,
                                                     (short)0, acc, false, false);
    }
    // C 16x16 f32: VGPR r -> row = subtile*16 + half*8 + r, col = nT*16 + l16
    const int col = nColG + l16;
    const float bv = bias ? bias[col] : 0.f;
    const int rowBase = mT * 64 + wave * 16 + half * 8;
    #pragma unroll
    for (int r = 0; r < 8; ++r) {
        int row = rowBase + r;
        long idx = (long)row * ldc + col;
        float v = acc[r] * alpha + bv;
        if (Cres) v += Cres[idx];
        C[idx] = v;
    }
}

// ------------------------------------------------------------- small utils
__global__ void k_cvt_f16(const float* __restrict__ in, f16* __restrict__ out, int n) {
    int i = blockIdx.x * blockDim.x + threadIdx.x;
    if (i < n) out[i] = (f16)in[i];
}

// out f16 (C x R) = transpose of in f32 (R x C, row stride ldin); batched over z.
__global__ void k_transpose_f16(const float* in, f16* out,
                                int R, int C, int ldin, int H,
                                long inOffB, long inOffH, long outOffZ) {
    int z = blockIdx.z;
    in  += (z / H) * inOffB + (z % H) * inOffH;
    out += (long)z * outOffZ;
    int idx = blockIdx.x * blockDim.x + threadIdx.x;
    if (idx >= R * C) return;
    int r = idx / C, c = idx % C;
    out[(long)c * R + r] = (f16)in[(long)r * ldin + c];
}

__global__ void k_copy(const float* __restrict__ in, float* __restrict__ out, int n) {
    int i = blockIdx.x * blockDim.x + threadIdx.x;
    if (i < n) out[i] = in[i];
}

// im2col for k=3 pad=1 time conv, input (B,F,256) -> rows (B*F) x 768,
// ordering j = chan*3 + tap to match OIH weight flattening.
__global__ void k_im2col3(const float* __restrict__ in, f16* __restrict__ out) {
    int i = blockIdx.x * blockDim.x + threadIdx.x;
    if (i >= BF * 768) return;
    int row = i / 768, j = i % 768;
    int c = j / 3, h = j % 3;
    int b = row / FF, f = row % FF;
    int fs = f + h - 1;
    float v = (fs >= 0 && fs < FF) ? in[((long)b * FF + fs) * DM + c] : 0.f;
    out[i] = (f16)v;
}

// -------------------------------------------------------------- norm layers
__global__ __launch_bounds__(512)
void k_groupnorm_leaky(const float* __restrict__ in, float* __restrict__ out,
                       const float* __restrict__ s, const float* __restrict__ bp) {
    int b = blockIdx.x / 4, g = blockIdx.x % 4;    // 4 groups of 64 channels
    int tid = threadIdx.x;
    __shared__ float sh[512], sh2[512];
    float sum = 0.f, sq = 0.f;
    for (int it = 0; it < 64; ++it) {
        int e = it * 512 + tid;                    // 32768 elems per (b,g)
        int f = e >> 6, c = g * 64 + (e & 63);
        float v = in[((long)b * FF + f) * DM + c];
        sum += v; sq += v * v;
    }
    sh[tid] = sum; sh2[tid] = sq; __syncthreads();
    for (int d = 256; d > 0; d >>= 1) {
        if (tid < d) { sh[tid] += sh[tid + d]; sh2[tid] += sh2[tid + d]; }
        __syncthreads();
    }
    float mu  = sh[0] * (1.f / 32768.f);
    float var = sh2[0] * (1.f / 32768.f) - mu * mu;
    float inv = rsqrtf(var + 1e-5f);
    for (int it = 0; it < 64; ++it) {
        int e = it * 512 + tid;
        int f = e >> 6, c = g * 64 + (e & 63);
        long idx = ((long)b * FF + f) * DM + c;
        float v = (in[idx] - mu) * inv * s[c] + bp[c];
        out[idx] = v > 0.f ? v : 0.01f * v;        // leaky relu
    }
}

__global__ __launch_bounds__(256)
void k_layernorm(const float* __restrict__ in, float* __restrict__ out,
                 const float* __restrict__ s, const float* __restrict__ bp) {
    int row = blockIdx.x, c = threadIdx.x;
    __shared__ float sh[256], sh2[256];
    float v = in[(long)row * DM + c];
    sh[c] = v; sh2[c] = v * v; __syncthreads();
    for (int d = 128; d > 0; d >>= 1) {
        if (c < d) { sh[c] += sh[c + d]; sh2[c] += sh2[c + d]; }
        __syncthreads();
    }
    float mu = sh[0] * (1.f / 256.f), var = sh2[0] * (1.f / 256.f) - mu * mu;
    out[(long)row * DM + c] = (v - mu) * rsqrtf(var + 1e-5f) * s[c] + bp[c];
}

__global__ void k_embed(float* __restrict__ x, const float* __restrict__ f0,
                        const float* __restrict__ vol, const float* __restrict__ pf,
                        const float* f0w, const float* f0b,
                        const float* phw, const float* phb,
                        const float* vw,  const float* vb) {
    int i = blockIdx.x * blockDim.x + threadIdx.x;
    if (i >= BF * DM) return;
    int row = i / DM, d = i % DM;
    float l = log1pf(f0[row] * (1.f / 700.f));
    float p = pf[row] * (1.f / PI_F);
    x[i] += l * f0w[d] + f0b[d] + p * phw[d] + phb[d] + vol[row] * vw[d] + vb[d];
}

__global__ __launch_bounds__(512)
void k_softmax512(float* __restrict__ x) {
    long row = blockIdx.x; int t = threadIdx.x;
    __shared__ float sh[512];
    float v = x[row * 512 + t];
    sh[t] = v; __syncthreads();
    for (int d = 256; d > 0; d >>= 1) { if (t < d) sh[t] = fmaxf(sh[t], sh[t + d]); __syncthreads(); }
    float m = sh[0]; __syncthreads();
    float e = __expf(v - m);
    sh[t] = e; __syncthreads();
    for (int d = 256; d > 0; d >>= 1) { if (t < d) sh[t] += sh[t + d]; __syncthreads(); }
    x[row * 512 + t] = e / sh[0];
}

__global__ void k_glu(const float* __restrict__ in, float* __restrict__ out) {
    int i = blockIdx.x * blockDim.x + threadIdx.x;
    if (i >= BF * 512) return;
    long row = i / 512; int c = i % 512;
    float a = in[row * 1024 + c], g = in[row * 1024 + 512 + c];
    out[i] = a * (1.f / (1.f + __expf(-g)));
}

__global__ void k_dwconv_swish(const float* __restrict__ in, float* __restrict__ out,
                               const float* __restrict__ w, const float* __restrict__ bp) {
    int i = blockIdx.x * blockDim.x + threadIdx.x;
    if (i >= BF * 512) return;
    long row = i / 512; int c = i % 512;
    int b = (int)(row / FF), f = (int)(row % FF);
    float acc = bp[c];
    for (int j = 0; j < 31; ++j) {
        int fs = f + j - 15;
        if (fs >= 0 && fs < FF) acc += in[((long)b * FF + fs) * 512 + c] * w[c * 31 + j];
    }
    out[i] = acc * (1.f / (1.f + __expf(-acc)));   // swish
}

// ------------------------------------------------------- phase / cumsum path
__global__ void k_f0_upsample(const float* __restrict__ f0, float* __restrict__ f0s) {
    int i = blockIdx.x * blockDim.x + threadIdx.x;
    if (i >= BTOT) return;
    int b = i / TT, t = i % TT;
    int j = t / BLK;
    float fr = (float)(t % BLK) * (1.f / BLK);
    int j1 = j + 1 > FF - 1 ? FF - 1 : j + 1;      // edge-repeat
    float s0 = f0[b * FF + j], s1 = f0[b * FF + j1];
    f0s[i] = s0 * (1.f - fr) + s1 * fr;
}

__global__ __launch_bounds__(256)
void k_framesum(const float* __restrict__ f0s, float* __restrict__ fsum) {
    int row = blockIdx.x, tid = threadIdx.x;
    __shared__ float sh[256];
    long base = (long)row * BLK;
    sh[tid] = (f0s[base + tid] + f0s[base + 256 + tid]) * (1.f / 44100.f);
    __syncthreads();
    for (int d = 128; d > 0; d >>= 1) { if (tid < d) sh[tid] += sh[tid + d]; __syncthreads(); }
    if (tid == 0) fsum[row] = sh[0];
}

__global__ __launch_bounds__(512)
void k_framescan(const float* __restrict__ fsum, float* __restrict__ foff) {
    int b = blockIdx.x, t = threadIdx.x;
    __shared__ float sh[512];
    float v = fsum[b * FF + t];
    sh[t] = v; __syncthreads();
    for (int d = 1; d < 512; d <<= 1) {
        float a = (t >= d) ? sh[t - d] : 0.f; __syncthreads();
        sh[t] += a; __syncthreads();
    }
    foff[b * FF + t] = sh[t] - v;                  // exclusive prefix
}

__global__ __launch_bounds__(512)
void k_phase(const float* __restrict__ f0s, const float* __restrict__ foff,
             float* __restrict__ phase, float* __restrict__ pf) {
    int row = blockIdx.x, t = threadIdx.x;
    __shared__ float sh[512];
    long base = (long)row * BLK;
    float v = f0s[base + t] * (1.f / 44100.f);
    sh[t] = v; __syncthreads();
    for (int d = 1; d < 512; d <<= 1) {
        float a = (t >= d) ? sh[t - d] : 0.f; __syncthreads();
        sh[t] += a; __syncthreads();
    }
    float cum = sh[t] + foff[row];
    float xw = cum - rintf(cum);                   // round-half-even like jnp.round
    float ph = 2.f * PI_F * xw;
    phase[base + t] = ph;
    if (t == 0) pf[row] = ph;                      // phase_frames
}

// ------------------------------------------------------------- synth + ctrl
__global__ void k_amp(const float* __restrict__ e, const float* __restrict__ f0,
                      float* __restrict__ amp) {
    int i = blockIdx.x * blockDim.x + threadIdx.x;
    if (i >= BF * NHARM) return;
    int row = i / NHARM, n = i % NHARM;
    float a = __expf(e[(long)row * 640 + n]) * (1.f / 128.f);
    float mask = (f0[row] * (float)(n + 1) < 22050.f) ? (1.f + 1e-7f) : 1e-7f;
    amp[i] = a * mask;
}

__global__ __launch_bounds__(256)
void k_gdscan(const float* __restrict__ e, float* __restrict__ psi) {
    int row = blockIdx.x, t = threadIdx.x;
    __shared__ float sh[256];
    float v = PI_F * tanhf(e[(long)row * 640 + NHARM + t]);
    sh[t] = v; __syncthreads();
    for (int d = 1; d < 256; d <<= 1) {
        float a = (t >= d) ? sh[t - d] : 0.f; __syncthreads();
        sh[t] += a; __syncthreads();
    }
    psi[(long)row * 256 + t] = sh[t];              // cumulative group delay phase
}

__global__ void k_nmag(const float* __restrict__ e, float* __restrict__ nm) {
    int i = blockIdx.x * blockDim.x + threadIdx.x;
    if (i >= BF * 256) return;
    int row = i / 256, j = i % 256;
    nm[i] = __expf(e[(long)row * 640 + 384 + j]) * (1.f / 128.f);
}

__global__ __launch_bounds__(512)
void k_synth(const float* __restrict__ phase, const float* __restrict__ amp,
             float* __restrict__ sinus) {
    int row = blockIdx.x;                          // b*FF + f
    int b = row / FF, f = row % FF;
    int t = threadIdx.x;
    __shared__ float a0[NHARM], a1[NHARM];
    int f1 = f + 1 > FF - 1 ? FF - 1 : f + 1;
    if (t < NHARM) {
        a0[t] = amp[((long)b * FF + f)  * NHARM + t];
        a1[t] = amp[((long)b * FF + f1) * NHARM + t];
    }
    __syncthreads();
    float fr = (float)t * (1.f / BLK);
    float ph = phase[(long)row * BLK + t];
    float s = 0.f;
    #pragma unroll 4
    for (int n = 0; n < NHARM; ++n) {
        float a = a0[n] * (1.f - fr) + a1[n] * fr;
        s += __sinf(ph * (float)(n + 1)) * a;      // transcendental-bound core
    }
    sinus[(long)row * BLK + t] = s;
}

// --------------------------------------------------------------- threefry2x32
__device__ __forceinline__ unsigned rotl32(unsigned x, int r) {
    return (x << r) | (x >> (32 - r));
}
__device__ void threefry2x32(unsigned k0, unsigned k1, unsigned c0, unsigned c1,
                             unsigned& o0, unsigned& o1) {
    const unsigned rot[8] = {13, 15, 26, 6, 17, 29, 16, 24};
    unsigned ks[3] = {k0, k1, 0x1BD11BDAu ^ k0 ^ k1};
    unsigned x0 = c0 + k0, x1 = c1 + k1;
    #pragma unroll
    for (int g = 0; g < 5; ++g) {
        #pragma unroll
        for (int i = 0; i < 4; ++i) {
            x0 += x1; x1 = rotl32(x1, rot[(g & 1) * 4 + i]); x1 ^= x0;
        }
        x0 += ks[(g + 1) % 3];
        x1 += ks[(g + 2) % 3] + (unsigned)(g + 1);
    }
    o0 = x0; o1 = x1;
}
__global__ void k_noise(float* __restrict__ noise) {
    int i = blockIdx.x * blockDim.x + threadIdx.x;
    const int half = BTOT / 2;
    if (i >= half) return;
    unsigned o0, o1;                               // jax.random.key(123) = (0,123)
    threefry2x32(0u, 123u, (unsigned)i, (unsigned)(i + half), o0, o1);
    float u0 = __uint_as_float((o0 >> 9) | 0x3F800000u) - 1.f;
    float u1 = __uint_as_float((o1 >> 9) | 0x3F800000u) - 1.f;
    noise[i]        = u0 * 2.f - 1.f;
    noise[i + half] = u1 * 2.f - 1.f;
}

// ------------------------------------------------- IR synthesis (cosine DFT)
// ir = roll(irfft(exp(i*psi), n=510), 255); rolled index tr=(t+255)%510.
__global__ __launch_bounds__(512)
void k_ir_harm(const float* __restrict__ psi, float* __restrict__ ir) {
    int row = blockIdx.x, t = threadIdx.x;
    __shared__ float p[256];
    if (t < 256) p[t] = psi[(long)row * 256 + t];
    __syncthreads();
    if (t >= LFILT) return;
    int tr = (t + 255) % LFILT;
    float th = (2.f * PI_F / LFILT) * (float)tr;
    float acc = __cosf(p[0]) + ((tr & 1) ? -1.f : 1.f) * __cosf(p[255]);
    for (int k = 1; k < 255; ++k) acc += 2.f * __cosf(p[k] + th * (float)k);
    ir[(long)row * LFILT + t] = acc * (1.f / LFILT);
}
__global__ __launch_bounds__(512)
void k_ir_noise(const float* __restrict__ nm, float* __restrict__ ir) {
    int row = blockIdx.x, t = threadIdx.x;
    __shared__ float p[256];
    if (t < 256) p[t] = nm[(long)row * 256 + t];
    __syncthreads();
    if (t >= LFILT) return;
    int tr = (t + 255) % LFILT;
    float th = (2.f * PI_F / LFILT) * (float)tr;
    float acc = p[0] + ((tr & 1) ? -1.f : 1.f) * p[255];
    for (int k = 1; k < 255; ++k) acc += 2.f * p[k] * __cosf(th * (float)k);
    float win = 0.5f - 0.5f * __cosf((2.f * PI_F / LFILT) * (float)t);  // hann, post-roll
    ir[(long)row * LFILT + t] = acc * win * (1.f / LFILT);
}

// Time-domain frame conv + overlap-add gather (== FFT path, n_fft=1024).
__global__ __launch_bounds__(512)
void k_filter_out(const float* __restrict__ sinus, const float* __restrict__ nraw,
                  const float* __restrict__ irH,  const float* __restrict__ irN,
                  float* __restrict__ out) {
    int gi = blockIdx.x * blockDim.x + threadIdx.x;
    if (gi >= BTOT) return;
    int b = gi / TT, t = gi % TT;
    float accH = 0.f, accN = 0.f;
    int fLo = (t - 765 + 511) / 512; if (fLo < 0) fLo = 0;       // n <= 1020
    int fHi = (t + 255) / 512;       if (fHi > FF - 1) fHi = FF - 1;
    for (int f = fLo; f <= fHi; ++f) {
        int n = t + 255 - 512 * f;
        int kLo = n - (LFILT - 1); if (kLo < 0) kLo = 0;
        int kHi = n < (BLK - 1) ? n : (BLK - 1);
        const float* fr  = sinus + (long)b * TT + (long)f * BLK;
        const float* fn  = nraw  + (long)b * TT + (long)f * BLK;
        const float* ih  = irH + ((long)b * FF + f) * LFILT;
        const float* inr = irN + ((long)b * FF + f) * LFILT;
        __builtin_prefetch(fr + kLo, 0, 3);        // global_prefetch_b8
        __builtin_prefetch(ih, 0, 3);
        __builtin_prefetch(fn + kLo, 0, 3);
        __builtin_prefetch(inr, 0, 3);
        for (int k = kLo; k <= kHi; ++k) {
            accH += fr[k] * ih[n - k];
            accN += fn[k] * inr[n - k];
        }
    }
    const long hidOff = (long)BF * DM;
    out[gi] = accH + accN;                           // signal
    out[BTOT + hidOff + gi] = accH;                  // harmonic
    out[BTOT + hidOff + BTOT + gi] = accN;           // noise
}

// -------------------------------------------------------------------- host
static void gemm(hipStream_t st, const f16* A, const f16* Bt, const float* bias,
                 const float* Cres, float* C, int M, int N, int K,
                 int lda, int ldb, int ldc, float alpha,
                 int Z = 1, int H = 1,
                 long aOffB = 0, long aOffH = 0, long bOffB = 0, long bOffH = 0,
                 long cOffB = 0, long cOffH = 0) {
    dim3 g((unsigned)(N / 16), (unsigned)(M / 64), (unsigned)Z);  // M % 64 == 0
    k_gemm_f16<<<g, 128, 0, st>>>(A, Bt, bias, Cres, C, M, N, K, lda, ldb, ldc,
                                  alpha, H, aOffB, aOffH, bOffB, bOffH, cOffB, cOffH);
}
static inline int cdiv(long n, int b) { return (int)((n + b - 1) / b); }

extern "C" void kernel_launch(void* const* d_in, const int* in_sizes, int n_in,
                              void* d_out, int out_size, void* d_ws, size_t ws_size,
                              hipStream_t stream) {
    if (n_in < 61) return;
    const float* units   = (const float*)d_in[0];
    const float* f0      = (const float*)d_in[1];
    const float* vol     = (const float*)d_in[2];
    const float* conv1_b = (const float*)d_in[3];
    const float* conv1_w = (const float*)d_in[4];
    const float* conv2_b = (const float*)d_in[5];
    const float* conv2_w = (const float*)d_in[6];
    const float* f0_b    = (const float*)d_in[7];
    const float* f0_w    = (const float*)d_in[8];
    const float* fin_b   = (const float*)d_in[9];
    const float* fin_s   = (const float*)d_in[10];
    const float* gn_b    = (const float*)d_in[11];
    const float* gn_s    = (const float*)d_in[12];
    const float* out_b   = (const float*)d_in[55];
    const float* out_w   = (const float*)d_in[56];
    const float* ph_b    = (const float*)d_in[57];
    const float* ph_w    = (const float*)d_in[58];
    const float* vol_b   = (const float*)d_in[59];
    const float* vol_w   = (const float*)d_in[60];

    char* w = (char*)d_ws;                         // workspace layout (MB offsets)
    float* f0s    = (float*)(w + (0L  << 20));     // 4MB
    float* phase  = (float*)(w + (4L  << 20));     // 4MB
    float* sinus  = (float*)(w + (8L  << 20));     // 4MB
    float* nraw   = (float*)(w + (12L << 20));     // 4MB
    float* x      = (float*)(w + (16L << 20));     // 2MB residual stream
    float* xln    = (float*)(w + (18L << 20));     // 2MB layernorm out
    float* big    = (float*)(w + (20L << 20));     // 8MB (conv/qkv/pw1/e)
    float* big2   = (float*)(w + (28L << 20));     // 8MB (gn out / glu / dw)
    f16*   A16    = (f16*)  (w + (36L << 20));     // 8MB f16 staging (4M elems)
    f16*   B16    = (f16*)  (w + (44L << 20));     // 2MB f16 weights/vT
    float* scores = (float*)(w + (46L << 20));     // 8MB per-batch attention
    float* amp    = (float*)(w + (54L << 20));     // 1MB
    float* psi    = (float*)(w + (55L << 20));     // 2MB
    float* nmag   = (float*)(w + (57L << 20));     // 2MB
    float* irH    = (float*)(w + (59L << 20));     // 4.2MB
    float* irN    = (float*)(w + (64L << 20));     // 4.2MB
    float* fsum   = (float*)(w + (69L << 20));
    float* foff   = fsum + BF;
    float* pf     = foff + BF;
    float* attnO  = (float*)(w + (70L << 20));     // 2MB
    f16* x16   = A16;                              // 0.5M f16 elems
    f16* qkv16 = A16 + (512 * 1024);               // 1.5M elems
    f16* sc16  = A16 + (2 * 1024 * 1024);          // 2.0M elems
    f16* vT    = B16 + (512 * 1024);

    // ---- phase path: upsample f0, hierarchical cumsum aligned to frames
    k_f0_upsample<<<cdiv(BTOT, 256), 256, 0, stream>>>(f0, f0s);
    k_framesum  <<<BF, 256, 0, stream>>>(f0s, fsum);
    k_framescan <<<BB, 512, 0, stream>>>(fsum, foff);
    k_phase     <<<BF, 512, 0, stream>>>(f0s, foff, phase, pf);

    // ---- frontend: conv1(k3) -> GN+leaky -> conv2(k3) -> embeddings
    k_im2col3<<<cdiv((long)BF * 768, 256), 256, 0, stream>>>(units, A16);
    k_cvt_f16<<<cdiv(256 * 768, 256), 256, 0, stream>>>(conv1_w, B16, 256 * 768);
    gemm(stream, A16, B16, conv1_b, nullptr, big, BF, DM, 768, 768, 768, DM, 1.f);
    k_groupnorm_leaky<<<BB * 4, 512, 0, stream>>>(big, big2, gn_s, gn_b);
    k_im2col3<<<cdiv((long)BF * 768, 256), 256, 0, stream>>>(big2, A16);
    k_cvt_f16<<<cdiv(256 * 768, 256), 256, 0, stream>>>(conv2_w, B16, 256 * 768);
    gemm(stream, A16, B16, conv2_b, nullptr, x, BF, DM, 768, 768, 768, DM, 1.f);
    k_embed<<<cdiv(BF * DM, 256), 256, 0, stream>>>(x, f0, vol, pf, f0_w, f0_b,
                                                    ph_w, ph_b, vol_w, vol_b);

    // ---- 3 conformer layers
    for (int l = 0; l < 3; ++l) {
        const float* cln_b = (const float*)d_in[13 + 14 * l + 0];
        const float* cln_s = (const float*)d_in[13 + 14 * l + 1];
        const float* dw_b  = (const float*)d_in[13 + 14 * l + 2];
        const float* dw_w  = (const float*)d_in[13 + 14 * l + 3];
        const float* ln1_b = (const float*)d_in[13 + 14 * l + 4];
        const float* ln1_s = (const float*)d_in[13 + 14 * l + 5];
        const float* o_b   = (const float*)d_in[13 + 14 * l + 6];
        const float* o_w   = (const float*)d_in[13 + 14 * l + 7];
        const float* pw1_b = (const float*)d_in[13 + 14 * l + 8];
        const float* pw1_w = (const float*)d_in[13 + 14 * l + 9];
        const float* pw2_b = (const float*)d_in[13 + 14 * l + 10];
        const float* pw2_w = (const float*)d_in[13 + 14 * l + 11];
        const float* qkv_b = (const float*)d_in[13 + 14 * l + 12];
        const float* qkv_w = (const float*)d_in[13 + 14 * l + 13];

        // attention: x + Attn(LN(x))
        k_layernorm<<<BF, 256, 0, stream>>>(x, xln, ln1_s, ln1_b);
        k_cvt_f16<<<cdiv(BF * DM, 256), 256, 0, stream>>>(xln, x16, BF * DM);
        k_transpose_f16<<<dim3(cdiv(256 * 768, 256), 1, 1), 256, 0, stream>>>(
            qkv_w, B16, 256, 768, 768, 1, 0, 0, 0);
        gemm(stream, x16, B16, qkv_b, nullptr, big, BF, 768, DM, DM, DM, 768, 1.f);
        k_cvt_f16<<<cdiv((long)BF * 768, 256), 256, 0, stream>>>(big, qkv16, BF * 768);
        for (int b = 0; b < BB; ++b) {
            long qb = (long)b * FF * 768;
            // scores = softmax(q k^T / sqrt(32)) per head (z = head)
            gemm(stream, qkv16 + qb, qkv16 + qb + 256, nullptr, nullptr, scores,
                 FF, FF, 32, 768, 768, FF, 0.17677669529663687f,
                 8, 8, 0, 32, 0, 32, 0, (long)FF * FF);
            k_softmax512<<<8 * FF, 512, 0, stream>>>(scores);
            k_cvt_f16<<<cdiv((long)8 * FF * FF, 256), 256, 0, stream>>>(
                scores, sc16, 8 * FF * FF);
            k_transpose_f16<<<dim3(cdiv(FF * 32, 256), 1, 8), 256, 0, stream>>>(
                big + qb + 512, vT, FF, 32, 768, 8, 0, 32, (long)32 * FF);
            gemm(stream, sc16, vT, nullptr, nullptr, attnO + (long)b * FF * DM,
                 FF, 32, FF, FF, FF, DM, 1.f,
                 8, 8, 0, (long)FF * FF, 0, (long)32 * FF, 0, 32);
        }
        k_cvt_f16<<<cdiv(BF * DM, 256), 256, 0, stream>>>(attnO, x16, BF * DM);
        k_transpose_f16<<<dim3(cdiv(256 * 256, 256), 1, 1), 256, 0, stream>>>(
            o_w, B16, 256, 256, 256, 1, 0, 0, 0);
        gemm(stream, x16, B16, o_b, x, x, BF, DM, DM, DM, DM, DM, 1.f);  // +residual

        // conformer conv: x + PW2(Swish(DW(GLU(PW1(LN(x))))))
        k_layernorm<<<BF, 256, 0, stream>>>(x, xln, cln_s, cln_b);
        k_cvt_f16<<<cdiv(BF * DM, 256), 256, 0, stream>>>(xln, x16, BF * DM);
        k_cvt_f16<<<cdiv(1024 * 256, 256), 256, 0, stream>>>(pw1_w, B16, 1024 * 256);
        gemm(stream, x16, B16, pw1_b, nullptr, big, BF, 1024, DM, DM, DM, 1024, 1.f);
        k_glu<<<cdiv(BF * 512, 256), 256, 0, stream>>>(big, big2);
        k_dwconv_swish<<<cdiv(BF * 512, 256), 256, 0, stream>>>(
            big2, big2 + (long)BF * 512, dw_w, dw_b);
        k_cvt_f16<<<cdiv(BF * 512, 256), 256, 0, stream>>>(
            big2 + (long)BF * 512, x16, BF * 512);
        k_cvt_f16<<<cdiv(256 * 512, 256), 256, 0, stream>>>(pw2_w, B16, 256 * 512);
        gemm(stream, x16, B16, pw2_b, x, x, BF, DM, 512, 512, 512, DM, 1.f);
    }

    // ---- final LN (= hidden output) + output projection (640)
    k_layernorm<<<BF, 256, 0, stream>>>(x, xln, fin_s, fin_b);
    k_cvt_f16<<<cdiv(BF * DM, 256), 256, 0, stream>>>(xln, x16, BF * DM);
    k_transpose_f16<<<dim3(cdiv(256 * 640, 256), 1, 1), 256, 0, stream>>>(
        out_w, B16, 256, 640, 640, 1, 0, 0, 0);
    gemm(stream, x16, B16, out_b, nullptr, big, BF, 640, DM, DM, DM, 640, 1.f);

    // ---- controls -> synthesis -> filtering -> outputs
    k_amp   <<<cdiv(BF * NHARM, 256), 256, 0, stream>>>(big, f0, amp);
    k_gdscan<<<BF, 256, 0, stream>>>(big, psi);
    k_nmag  <<<cdiv(BF * 256, 256), 256, 0, stream>>>(big, nmag);
    k_synth <<<BF, 512, 0, stream>>>(phase, amp, sinus);
    k_noise <<<cdiv(BTOT / 2, 256), 256, 0, stream>>>(nraw);
    k_ir_harm <<<BF, 512, 0, stream>>>(psi, irH);
    k_ir_noise<<<BF, 512, 0, stream>>>(nmag, irN);
    k_filter_out<<<cdiv(BTOT, 512), 512, 0, stream>>>(sinus, nraw, irH, irN,
                                                      (float*)d_out);
    k_copy<<<cdiv(BF * DM, 256), 256, 0, stream>>>(xln, (float*)d_out + BTOT, BF * DM);
}
